// Block_38165079392391
// MI455X (gfx1250) — matrix operand order repaired and checked by reading, not compiled, not run on hardware
//
#include <hip/hip_runtime.h>
#include <hip/hip_fp16.h>
#include <hip/hip_bf16.h>

typedef __attribute__((ext_vector_type(16))) _Float16 v16h;
typedef __attribute__((ext_vector_type(8)))  _Float16 v8h;
typedef __attribute__((ext_vector_type(8)))  float    v8f;

namespace {

constexpr int Bn  = 128;   // batch
constexpr int Tn  = 256;   // sequence
constexpr int Cn  = 384;   // embedding
constexpr int Hn  = 8;     // heads
constexpr int HSn = 48;    // head size
constexpr int HSP = 64;    // padded head size (mult of 32 for WMMA K)
constexpr int BT  = Bn * Tn;
constexpr int FF  = 4 * Cn; // 1536

// ---------------------------------------------------------------------------
// Fragment loaders (wave32, 16x16x32 f16 WMMA), contiguous k order used by both A and B.
// A (16xK row-major, lda in halves): lane l&15 = row m; elements
//   i -> k = k0 + hi*16 + i
__device__ __forceinline__ v16h load_a_frag(const _Float16* A, int row0, int k0, int lda) {
  const int l  = threadIdx.x & 15;
  const int hi = (threadIdx.x >> 4) & 1;
  const _Float16* p = A + (size_t)(row0 + l) * lda + k0 + hi * 16;
  v8h lo = *(const v8h*)(p);
  v8h hh = *(const v8h*)(p + 8);
  v16h a;
#pragma unroll
  for (int i = 0; i < 8; ++i) { a[i] = lo[i]; a[i + 8] = hh[i]; }
  return a;
}

// B from Bt[N x K] row-major (Bt = B^T): lane l&15 = output col n;
// element i -> k = k0 + hi*16 + i => one contiguous 16-half (32B) run.
__device__ __forceinline__ v16h load_b_frag(const _Float16* Bt, int col0, int k0, int ldb) {
  const int l  = threadIdx.x & 15;
  const int hi = (threadIdx.x >> 4) & 1;
  const _Float16* p = Bt + (size_t)(col0 + l) * ldb + k0 + hi * 16;
  return *(const v16h*)p;
}

__device__ __forceinline__ v8f wmma_f16(v16h a, v16h b, v8f c) {
  return __builtin_amdgcn_wmma_f32_16x16x32_f16(false, a, false, b, (short)0, c,
                                                false, false);
}

// ---------------------------------------------------------------------------
// Weight transpose + f32->f16 convert (with zero row padding for HS->HSP).
__global__ void wtrans_kernel(const float* __restrict__ src, _Float16* __restrict__ dst,
                              int R, int Cc, int Cpad) {
  const size_t idx = (size_t)blockIdx.x * blockDim.x + threadIdx.x;
  if (idx >= (size_t)R * Cpad) return;
  const int n = (int)(idx / R);
  const int r = (int)(idx % R);
  const float* s = src + (size_t)blockIdx.z * R * Cc;
  _Float16* d = dst + (size_t)blockIdx.z * R * Cpad;
  d[(size_t)n * R + r] = (n < Cc) ? (_Float16)s[(size_t)r * Cc + n] : (_Float16)0.f;
}

// Fill the V^T pad row d = HSn (=48) with ones: the P x V WMMA then accumulates
// the softmax denominator in output column 48 for free (online-softmax trick).
__global__ void vones_kernel(_Float16* __restrict__ vT) {
  const size_t idx = (size_t)blockIdx.x * blockDim.x + threadIdx.x;  // bh*Tn + t
  const size_t bh = idx >> 8;
  const size_t t  = idx & 255;
  vT[(bh * HSP + HSn) * Tn + t] = (_Float16)1.f;
}

// ---------------------------------------------------------------------------
// LayerNorm fp32 -> f16, one wave per token row (C = 384 = 32 lanes * 12).
__global__ void ln_kernel(const float* __restrict__ x, const float* __restrict__ g,
                          const float* __restrict__ be, _Float16* __restrict__ out) {
  const int row  = blockIdx.x * (blockDim.x >> 5) + (threadIdx.x >> 5);
  const int lane = threadIdx.x & 31;
  const float* xr = x + (size_t)row * Cn;
  float v[12];
  float s = 0.f;
#pragma unroll
  for (int i = 0; i < 12; ++i) { v[i] = xr[lane + 32 * i]; s += v[i]; }
#pragma unroll
  for (int off = 1; off < 32; off <<= 1) s += __shfl_xor(s, off, 32);
  const float mu = s * (1.f / Cn);
  float var = 0.f;
#pragma unroll
  for (int i = 0; i < 12; ++i) { float d = v[i] - mu; var += d * d; }
#pragma unroll
  for (int off = 1; off < 32; off <<= 1) var += __shfl_xor(var, off, 32);
  const float rs = rsqrtf(var * (1.f / Cn) + 1e-5f);
  _Float16* orow = out + (size_t)row * Cn;
#pragma unroll
  for (int i = 0; i < 12; ++i) {
    const int col = lane + 32 * i;
    orow[col] = (_Float16)((v[i] - mu) * rs * g[col] + be[col]);
  }
}

// ---------------------------------------------------------------------------
// WMMA GEMM: C[M,N] = A[M,K] * Bt[N,K]^T , one 32x64 tile per wave
// (2 A-frags reuse each of 4 B-frags -> 8 WMMAs / 12 b128 loads per K-step).
// EPI 0: f16 store        (ldc, + blockIdx.z*strideC)
// EPI 1: f16 V-transpose store -> [z][b][d][t]  (Tn/Bn hardcoded)
// EPI 2: f32 store = acc + bias[n] + resid[m*ldc+n]
// EPI 3: f16 store = relu(acc + bias[n])
template <int EPI>
__global__ void gemm16(const _Float16* __restrict__ A, const _Float16* __restrict__ Bt,
                       const float* __restrict__ bias, const float* __restrict__ resid,
                       void* __restrict__ Cout,
                       int M, int N, int K, int lda, int ldb, int ldc,
                       long long strideB, long long strideC) {
  const int wave = threadIdx.x >> 5;
  const int row0 = (blockIdx.x * (blockDim.x >> 5) + wave) * 32;
  if (row0 >= M) return;
  const int col0 = blockIdx.y * 64;
  const _Float16* Bz = Bt + (size_t)blockIdx.z * strideB;

  v8f zero = {0.f, 0.f, 0.f, 0.f, 0.f, 0.f, 0.f, 0.f};
  v8f acc[2][4] = {{zero, zero, zero, zero}, {zero, zero, zero, zero}};

  for (int k0 = 0; k0 < K; k0 += 32) {
    if (k0 + 32 < K) {  // prefetch next A rows (global_prefetch_b8)
      __builtin_prefetch(A + (size_t)(row0 + (threadIdx.x & 15)) * lda + k0 + 32, 0, 0);
      __builtin_prefetch(A + (size_t)(row0 + 16 + (threadIdx.x & 15)) * lda + k0 + 32, 0, 0);
    }
    v16h a0 = load_a_frag(A, row0, k0, lda);
    v16h a1 = load_a_frag(A, row0 + 16, k0, lda);
#pragma unroll
    for (int j = 0; j < 4; ++j) {
      v16h b = load_b_frag(Bz, col0 + j * 16, k0, ldb);
      acc[0][j] = wmma_f16(a0, b, acc[0][j]);
      acc[1][j] = wmma_f16(a1, b, acc[1][j]);
    }
  }

  const int l  = threadIdx.x & 15;
  const int hi = (threadIdx.x >> 4) & 1;
#pragma unroll
  for (int mi = 0; mi < 2; ++mi) {
#pragma unroll
    for (int j = 0; j < 4; ++j) {
      const int n = col0 + j * 16 + l;
#pragma unroll
      for (int r = 0; r < 8; ++r) {
        const int m  = row0 + mi * 16 + hi * 8 + r;  // D layout: row = hi*8+r, col = l
        const float v = acc[mi][j][r];
        if (EPI == 0) {
          ((_Float16*)Cout + (size_t)blockIdx.z * strideC)[(size_t)m * ldc + n] =
              (_Float16)v;
        } else if (EPI == 1) {
          const int bb = m >> 8;              // m / Tn
          const int tl = m & 255;             // m % Tn
          ((_Float16*)Cout)[(((size_t)blockIdx.z * Bn + bb) * HSP + n) * Tn + tl] =
              (_Float16)v;
        } else if (EPI == 2) {
          ((float*)Cout)[(size_t)m * ldc + n] =
              v + bias[n] + resid[(size_t)m * ldc + n];
        } else {
          const float t = v + bias[n];
          ((_Float16*)Cout)[(size_t)m * ldc + n] = (_Float16)(t > 0.f ? t : 0.f);
        }
      }
    }
  }
}

// ---------------------------------------------------------------------------
// Flash-style causal attention. One wave per (bh, 16-row Q tile).
// Q,K: [H*B][T][HSP] f16 ; VT: [H*B][HSP][T] f16 (row 48 == ones) ;
// Out: [B*T][C] f16 (head-concat). Softmax denominator rides in O column 48.
__global__ void attn_kernel(const _Float16* __restrict__ Q, const _Float16* __restrict__ K,
                            const _Float16* __restrict__ VT, _Float16* __restrict__ Out) {
  const int t0   = blockIdx.x * 16;
  const int bh   = blockIdx.y;       // = head*Bn + b
  const int head = bh >> 7;          // / Bn
  const int b    = bh & 127;         // % Bn
  const _Float16* Qb = Q  + (size_t)bh * Tn * HSP;
  const _Float16* Kb = K  + (size_t)bh * Tn * HSP;
  const _Float16* Vb = VT + (size_t)bh * HSP * Tn;
  const int l  = threadIdx.x & 15;
  const int hi = (threadIdx.x >> 4) & 1;

  __shared__ alignas(64) _Float16 pbuf[16 * 32];

  const v16h aq0 = load_a_frag(Qb, t0, 0, HSP);
  const v16h aq1 = load_a_frag(Qb, t0, 32, HSP);

  v8f zero = {0.f, 0.f, 0.f, 0.f, 0.f, 0.f, 0.f, 0.f};
  v8f o[4] = {zero, zero, zero, zero};
  float mrow[8];
#pragma unroll
  for (int r = 0; r < 8; ++r) mrow[r] = -1e30f;

  const float scale = 0.14433756729740643f;  // 48^-0.5

  const int nchunk = (t0 + 16 + 31) >> 5;    // s-chunks of 32 needed for causal rows
  for (int c = 0; c < nchunk; ++c) {
    const int s0 = c * 32;
    v8f sc0 = zero, sc1 = zero;
    sc0 = wmma_f16(aq0, load_b_frag(Kb, s0,      0,  HSP), sc0);
    sc0 = wmma_f16(aq1, load_b_frag(Kb, s0,      32, HSP), sc0);
    sc1 = wmma_f16(aq0, load_b_frag(Kb, s0 + 16, 0,  HSP), sc1);
    sc1 = wmma_f16(aq1, load_b_frag(Kb, s0 + 16, 32, HSP), sc1);

    float p0[8], p1[8], corr[8];
#pragma unroll
    for (int r = 0; r < 8; ++r) {
      const int trow = t0 + hi * 8 + r;                 // global query index
      float e0 = (s0 + l      <= trow) ? sc0[r] * scale : -1e30f;
      float e1 = (s0 + 16 + l <= trow) ? sc1[r] * scale : -1e30f;
      float tm = fmaxf(e0, e1);
#pragma unroll
      for (int off = 1; off < 16; off <<= 1) tm = fmaxf(tm, __shfl_xor(tm, off, 32));
      const float mnew = fmaxf(mrow[r], tm);
      corr[r] = __expf(mrow[r] - mnew);
      p0[r]   = __expf(e0 - mnew);
      p1[r]   = __expf(e1 - mnew);
      mrow[r] = mnew;
    }
    // rescale running output accumulators (incl. denominator column 48)
#pragma unroll
    for (int j = 0; j < 4; ++j)
#pragma unroll
      for (int r = 0; r < 8; ++r) o[j][r] *= corr[r];

    // stage P (16x32) through LDS: D-layout -> A-layout reshaping
    __syncthreads();
#pragma unroll
    for (int r = 0; r < 8; ++r) {
      pbuf[(hi * 8 + r) * 32 + l]      = (_Float16)p0[r];
      pbuf[(hi * 8 + r) * 32 + 16 + l] = (_Float16)p1[r];
    }
    __syncthreads();
    const v16h ap = load_a_frag(pbuf, 0, 0, 32);
#pragma unroll
    for (int j = 0; j < 4; ++j)
      o[j] = wmma_f16(ap, load_b_frag(Vb, j * 16, s0, Tn), o[j]);
  }

  // denominator for rows hi*8+r lives in o[3] element r of lane (hi*16) (d==48)
  float rdenom[8];
#pragma unroll
  for (int r = 0; r < 8; ++r) {
    const float s = __shfl(o[3][r], threadIdx.x & 16, 32);
    rdenom[r] = 1.f / s;
  }

  // write concat-head output, dropping pad columns d >= 48
#pragma unroll
  for (int j = 0; j < 4; ++j) {
    const int d = j * 16 + l;
    if (d < HSn) {
#pragma unroll
      for (int r = 0; r < 8; ++r) {
        const int t = t0 + hi * 8 + r;
        Out[((size_t)(b * Tn + t)) * Cn + head * HSn + d] =
            (_Float16)(o[j][r] * rdenom[r]);
      }
    }
  }
}

}  // namespace

// ---------------------------------------------------------------------------
extern "C" void kernel_launch(void* const* d_in, const int* in_sizes, int n_in,
                              void* d_out, int out_size, void* d_ws, size_t ws_size,
                              hipStream_t stream) {
  const float* x     = (const float*)d_in[0];
  const float* Wq    = (const float*)d_in[1];
  const float* Wk    = (const float*)d_in[2];
  const float* Wv    = (const float*)d_in[3];
  const float* Wproj = (const float*)d_in[4];
  const float* bproj = (const float*)d_in[5];
  const float* W1    = (const float*)d_in[6];
  const float* b1    = (const float*)d_in[7];
  const float* W2    = (const float*)d_in[8];
  const float* b2    = (const float*)d_in[9];
  const float* g1    = (const float*)d_in[10];
  const float* be1   = (const float*)d_in[11];
  const float* g2    = (const float*)d_in[12];
  const float* be2   = (const float*)d_in[13];

  char* ws = (char*)d_ws;
  size_t off = 0;
  auto alloc = [&](size_t bytes) -> void* {
    off = (off + 255) & ~(size_t)255;
    void* p = ws + off;
    off += bytes;
    return p;
  };

  _Float16* hbuf  = (_Float16*)alloc((size_t)BT * Cn * 2);           // ln out (reused as h2)
  _Float16* qbuf  = (_Float16*)alloc((size_t)Hn * Bn * Tn * HSP * 2);
  _Float16* kbuf  = (_Float16*)alloc((size_t)Hn * Bn * Tn * HSP * 2);
  _Float16* vTbuf = (_Float16*)alloc((size_t)Hn * Bn * HSP * Tn * 2);
  _Float16* attnc = (_Float16*)alloc((size_t)BT * Cn * 2);
  float*    x2    = (float*)   alloc((size_t)BT * Cn * 4);
  _Float16* wqT   = (_Float16*)alloc((size_t)Hn * HSP * Cn * 2);
  _Float16* wkT   = (_Float16*)alloc((size_t)Hn * HSP * Cn * 2);
  _Float16* wvT   = (_Float16*)alloc((size_t)Hn * HSP * Cn * 2);
  _Float16* wpT   = (_Float16*)alloc((size_t)Cn * Cn * 2);
  _Float16* w1T   = (_Float16*)alloc((size_t)FF * Cn * 2);
  _Float16* w2T   = (_Float16*)alloc((size_t)Cn * FF * 2);
  _Float16* a1    = qbuf;  // FFN hidden aliases q/k/vT region (attention is done)

  // --- weight prep: transpose + f16 (+ zero-pad HS 48 -> 64) ---
  wtrans_kernel<<<dim3(96, 1, Hn), 256, 0, stream>>>(Wq, wqT, Cn, HSn, HSP);
  wtrans_kernel<<<dim3(96, 1, Hn), 256, 0, stream>>>(Wk, wkT, Cn, HSn, HSP);
  wtrans_kernel<<<dim3(96, 1, Hn), 256, 0, stream>>>(Wv, wvT, Cn, HSn, HSP);
  wtrans_kernel<<<dim3(576, 1, 1), 256, 0, stream>>>(Wproj, wpT, Cn, Cn, Cn);
  wtrans_kernel<<<dim3(2304, 1, 1), 256, 0, stream>>>(W1, w1T, Cn, FF, FF);
  wtrans_kernel<<<dim3(2304, 1, 1), 256, 0, stream>>>(W2, w2T, FF, Cn, Cn);

  // --- ln1 ---
  ln_kernel<<<dim3(BT / 8), 256, 0, stream>>>(x, g1, be1, hbuf);

  // --- QKV projections (batched over heads, N=64 padded) ---
  const long long sB = (long long)HSP * Cn;          // per-head Bt stride
  const long long sC = (long long)Bn * Tn * HSP;     // per-head out stride
  gemm16<0><<<dim3(256, 1, Hn), 128, 0, stream>>>(hbuf, wqT, nullptr, nullptr, qbuf,
                                                  BT, HSP, Cn, Cn, Cn, HSP, sB, sC);
  gemm16<0><<<dim3(256, 1, Hn), 128, 0, stream>>>(hbuf, wkT, nullptr, nullptr, kbuf,
                                                  BT, HSP, Cn, Cn, Cn, HSP, sB, sC);
  gemm16<1><<<dim3(256, 1, Hn), 128, 0, stream>>>(hbuf, wvT, nullptr, nullptr, vTbuf,
                                                  BT, HSP, Cn, Cn, Cn, HSP, sB, sC);

  // --- set V^T ones row (softmax denominator column) ---
  vones_kernel<<<dim3(Hn * Bn * Tn / 256), 256, 0, stream>>>(vTbuf);

  // --- causal flash attention ---
  attn_kernel<<<dim3(Tn / 16, Hn * Bn), 32, 0, stream>>>(qbuf, kbuf, vTbuf, attnc);

  // --- output projection + bias + residual -> x2 (fp32) ---
  gemm16<2><<<dim3(256, Cn / 64, 1), 128, 0, stream>>>(attnc, wpT, bproj, x, x2,
                                                       BT, Cn, Cn, Cn, Cn, Cn, 0, 0);

  // --- ln2 -> h2 ---
  ln_kernel<<<dim3(BT / 8), 256, 0, stream>>>(x2, g2, be2, hbuf);

  // --- FFN1: relu(h2*W1 + b1) -> f16 ---
  gemm16<3><<<dim3(256, FF / 64, 1), 128, 0, stream>>>(hbuf, w1T, b1, nullptr, a1,
                                                       BT, FF, Cn, Cn, Cn, FF, 0, 0);

  // --- FFN2: x2 + a1*W2 + b2 -> d_out (fp32) ---
  gemm16<2><<<dim3(256, Cn / 64, 1), 128, 0, stream>>>(a1, w2T, b2, x2, d_out,
                                                       BT, Cn, FF, FF, FF, Cn, 0, 0);
}
